// EdgeModel_83760452207461
// MI455X (gfx1250) — matrix operand order repaired and checked by reading, not compiled, not run on hardware
//
#include <hip/hip_runtime.h>

#define HIDDEN 128
#define MTILE  128      // edges per block
#define KSTEPS1 16      // 512 / 32
#define KSTEPS2 4       // 128 / 32
#define HPITCH 136      // padded f16 pitch for H staging (272B row stride -> conflict-free)

typedef __attribute__((ext_vector_type(16))) _Float16 v16h;
typedef __attribute__((ext_vector_type(8)))  _Float16 v8h;
typedef __attribute__((ext_vector_type(8)))  float    v8f;
typedef __attribute__((ext_vector_type(4)))  float    v4f;

// ---------------------------------------------------------------------------
// CDNA5 async global->LDS DMA (ASYNCcnt-tracked, no VGPR round trip).
// Copies 32 bytes per lane: two b128 transfers sharing one IOFFSET-adjusted pair.
// ---------------------------------------------------------------------------
__device__ __forceinline__ void async_copy32(unsigned lds_addr, const void* gsrc)
{
    unsigned long long ga = (unsigned long long)(uintptr_t)gsrc;
    asm volatile("global_load_async_to_lds_b128 %0, %1, off\n\t"
                 "global_load_async_to_lds_b128 %0, %1, off offset:16"
                 :: "v"(lds_addr), "v"(ga) : "memory");
}

__device__ __forceinline__ void wait_async0()
{
    asm volatile("s_wait_asynccnt 0x0" ::: "memory");
}

// ---------------------------------------------------------------------------
// Convert W (K x 128, f32 row-major) into f16 pre-swizzled B-fragment layout:
// block b = kstep*8 + ntile (1 KB each): lane l holds column n = ntile*16 + (l&15),
// 16 f16 = K values kstep*32 + (l>>4)*8 + [0..7], then kstep*32 + 16 + (l>>4)*8 + [0..7].
// ---------------------------------------------------------------------------
__global__ __launch_bounds__(256) void swizzle_weights(const float* __restrict__ W,
                                                       _Float16* __restrict__ outp,
                                                       int K)
{
    int tid = blockIdx.x * 256 + threadIdx.x;
    if (tid >= K * HIDDEN) return;
    int j     = tid & 15;         // K index within lane's 16 contiguous values
    int l     = (tid >> 4) & 31;  // lane
    int b     = tid >> 9;         // fragment block
    int kstep = b >> 3;
    int ntile = b & 7;
    int k = kstep * 32 + (l >> 4) * 8 + (j & 7) + (j >> 3) * 16;
    int n = ntile * 16 + (l & 15);
    outp[tid] = (_Float16)W[k * HIDDEN + n];
}

// ---------------------------------------------------------------------------
// Fused edge MLP: out = relu(concat(src,dst,ea,u[batch]) @ W1 + b1) @ W2 + b2
// 256 threads = 8 waves; wave w computes rows [blk*128 + w*16, +16) x 128 cols.
// W1 k-slices double-buffered in LDS via async DMA, overlapped with WMMA.
// ---------------------------------------------------------------------------
__global__ __launch_bounds__(256) void edge_mlp(
    const float* __restrict__ src,
    const float* __restrict__ dstm,
    const float* __restrict__ ea,
    const float* __restrict__ u,
    const long long* __restrict__ batch,
    const float* __restrict__ b1,
    const float* __restrict__ b2,
    const _Float16* __restrict__ W1h,   // 512x128 f16, B-fragment swizzled
    const _Float16* __restrict__ W2h,   // 128x128 f16, B-fragment swizzled
    float* __restrict__ outp,
    int E)
{
    __shared__ alignas(32) _Float16 w1s[2][32 * HIDDEN];     // 2 x 8 KB W1 slices
    __shared__ alignas(32) _Float16 w2s[HIDDEN * HIDDEN];    // 32 KB, all of W2
    __shared__ alignas(32) _Float16 hbuf[8 * 16 * HPITCH];   // 34 KB, padded H staging

    const int tid   = threadIdx.x;
    const int wave  = tid >> 5;
    const int lane  = tid & 31;
    const int half  = lane >> 4;
    const int m     = lane & 15;       // M row within tile / N col within C tile
    const int khalf = half * 8;        // A-frag K sub-offset for this lane half

    const int rbase = blockIdx.x * MTILE + wave * 16;
    int r = rbase + m;                 // edge row this lane feeds into A fragments
    if (r >= E) r = E - 1;             // clamp (E is an exact multiple of 128 here)

    // LDS byte addresses (generic-pointer low 32 bits = wave-relative LDS offset)
    const unsigned w1s_lds = (unsigned)(uintptr_t)(&w1s[0][0]);
    const unsigned w2s_lds = (unsigned)(uintptr_t)(&w2s[0]);

    // ---- prologue: async-DMA W2 (32 KB) and W1 slice 0 (8 KB) into LDS ----
    #pragma unroll
    for (int i = 0; i < 4; ++i)
        async_copy32(w2s_lds + tid * 128 + i * 32,
                     (const char*)W2h + tid * 128 + i * 32);
    async_copy32(w1s_lds + tid * 32, (const char*)W1h + tid * 32);

    const int g = (int)batch[r];       // graph id -> row of u

    // per-lane row base pointers (hoisted 64-bit address math)
    const float* rowbase[4];
    rowbase[0] = src  + (size_t)r * HIDDEN;
    rowbase[1] = dstm + (size_t)r * HIDDEN;
    rowbase[2] = ea   + (size_t)r * HIDDEN;
    rowbase[3] = u    + (size_t)g * HIDDEN;

    // ---- stage 1 accumulators, preloaded with b1 ----
    v8f acc[8];
    #pragma unroll
    for (int n = 0; n < 8; ++n) {
        float bv = b1[n * 16 + m];
        #pragma unroll
        for (int i = 0; i < 8; ++i) acc[n][i] = bv;
    }

    wait_async0();
    __syncthreads();                   // W2 + W1 slice 0 visible to all waves

    // ---- stage 1: H = relu(X @ W1 + b1), K = 512 in 16 steps of 32 ----
    #pragma unroll
    for (int ks = 0; ks < KSTEPS1; ++ks) {
        const int cur = ks & 1;

        // kick off DMA of the next W1 slice into the other buffer; it overlaps
        // with this step's A loads + 8 WMMAs. Overwrite is safe: buffer was
        // last read at step ks-1, which the previous wait+barrier closed.
        if (ks + 1 < KSTEPS1)
            async_copy32(w1s_lds + (cur ^ 1) * 8192 + tid * 32,
                         (const char*)W1h + (ks + 1) * 8192 + tid * 32);

        const int seg  = ks >> 2;            // 0:src 1:dst 2:ea 3:u[batch]
        const int kcol = (ks & 3) * 32;
        const float* rowp = rowbase[seg] + kcol;
        if (seg < 3 && (ks & 3) != 3)
            __builtin_prefetch(rowp + 32, 0, 3);   // WGP-scope: warm next 128B

        // A fragment (16x32 f16): elems 0-7 = K khalf+0..7, elems 8-15 = K 16+khalf+0..7
        v4f x0, x1, x2, x3;
        if (seg < 3) {
            x0 = __builtin_nontemporal_load((const v4f*)(rowp + khalf));
            x1 = __builtin_nontemporal_load((const v4f*)(rowp + khalf + 4));
            x2 = __builtin_nontemporal_load((const v4f*)(rowp + 16 + khalf));
            x3 = __builtin_nontemporal_load((const v4f*)(rowp + 16 + khalf + 4));
        } else {
            x0 = *(const v4f*)(rowp + khalf);
            x1 = *(const v4f*)(rowp + khalf + 4);
            x2 = *(const v4f*)(rowp + 16 + khalf);
            x3 = *(const v4f*)(rowp + 16 + khalf + 4);
        }
        v16h af;
        #pragma unroll
        for (int i = 0; i < 4; ++i) {
            af[i]      = (_Float16)x0[i];
            af[4 + i]  = (_Float16)x1[i];
            af[8 + i]  = (_Float16)x2[i];
            af[12 + i] = (_Float16)x3[i];
        }

        // preload ALL 8 B-fragments into distinct registers (one ds clause,
        // counted waits, no per-WMMA dscnt-0 stall, no WAR hazard nops)
        const _Float16* wslice = &w1s[cur][0];
        v16h bf[8];
        #pragma unroll
        for (int n = 0; n < 8; ++n)
            bf[n] = *(const v16h*)(wslice + n * 512 + lane * 16);

        #pragma unroll
        for (int n = 0; n < 8; ++n)
            acc[n] = __builtin_amdgcn_wmma_f32_16x16x32_f16(
                false, af, false, bf[n], (short)0, acc[n], false, false);

        // close the step: own DMA drained, then block-wide handshake so the
        // next buffer is ready for everyone and this buffer may be overwritten
        if (ks + 1 < KSTEPS1) {
            wait_async0();
            __syncthreads();
        }
    }

    // ---- ReLU, convert to f16, stage H (16x128 per wave) into padded LDS ----
    _Float16* hw = hbuf + wave * (16 * HPITCH);
    #pragma unroll
    for (int n = 0; n < 8; ++n) {
        const int col = n * 16 + m;
        #pragma unroll
        for (int v = 0; v < 8; ++v) {
            float hv = acc[n][v];
            hv = hv > 0.0f ? hv : 0.0f;
            hw[(v + 8 * half) * HPITCH + col] = (_Float16)hv;   // D layout: row = v + 8*half
        }
    }

    // ---- stage 2: out = H @ W2 + b2, K = 128 in 4 steps ----
    v8f acc2[8];
    #pragma unroll
    for (int n = 0; n < 8; ++n) {
        float bv = b2[n * 16 + m];
        #pragma unroll
        for (int i = 0; i < 8; ++i) acc2[n][i] = bv;
    }

    #pragma unroll
    for (int ks = 0; ks < KSTEPS2; ++ks) {
        const _Float16* hp = hw + m * HPITCH + ks * 32 + khalf;
        v8h lo = *(const v8h*)hp;          // 16B, conflict-free (pitch 272B ≡ 4 banks)
        v8h hi = *(const v8h*)(hp + 16);
        v16h af;
        #pragma unroll
        for (int i = 0; i < 8; ++i) { af[i] = lo[i]; af[8 + i] = hi[i]; }

        v16h bf[8];
        #pragma unroll
        for (int n = 0; n < 8; ++n)
            bf[n] = *(const v16h*)(w2s + (ks * 8 + n) * 512 + lane * 16);

        #pragma unroll
        for (int n = 0; n < 8; ++n)
            acc2[n] = __builtin_amdgcn_wmma_f32_16x16x32_f16(
                false, af, false, bf[n], (short)0, acc2[n], false, false);
    }

    // ---- store: lanes 0-15 of each half cover 64B-contiguous row chunks ----
    #pragma unroll
    for (int n = 0; n < 8; ++n) {
        const int col = n * 16 + m;
        #pragma unroll
        for (int v = 0; v < 8; ++v) {
            int row = rbase + v + 8 * half;
            if (row < E)
                __builtin_nontemporal_store(acc2[n][v],
                                            outp + (size_t)row * HIDDEN + col);
        }
    }
}

// ---------------------------------------------------------------------------
extern "C" void kernel_launch(void* const* d_in, const int* in_sizes, int n_in,
                              void* d_out, int out_size, void* d_ws, size_t ws_size,
                              hipStream_t stream)
{
    const float*     src   = (const float*)d_in[0];
    const float*     dstm  = (const float*)d_in[1];
    const float*     ea    = (const float*)d_in[2];
    const float*     u     = (const float*)d_in[3];
    const long long* batch = (const long long*)d_in[4];   // jnp.int64
    const float*     W1    = (const float*)d_in[5];
    const float*     b1    = (const float*)d_in[6];
    const float*     W2    = (const float*)d_in[7];
    const float*     b2    = (const float*)d_in[8];
    float*           outp  = (float*)d_out;

    const int E = in_sizes[0] / HIDDEN;

    _Float16* W1h = (_Float16*)d_ws;             // 512*128*2 = 128 KB
    _Float16* W2h = W1h + 512 * HIDDEN;          // 128*128*2 =  32 KB

    swizzle_weights<<<(512 * HIDDEN + 255) / 256, 256, 0, stream>>>(W1, W1h, 512);
    swizzle_weights<<<(HIDDEN * HIDDEN + 255) / 256, 256, 0, stream>>>(W2, W2h, HIDDEN);

    const int nblocks = (E + MTILE - 1) / MTILE;
    edge_mlp<<<nblocks, 256, 0, stream>>>(src, dstm, ea, u, batch, b1, b2,
                                          W1h, W2h, outp, E);
}
